// ResVertixRefineShapenet_75952201662788
// MI455X (gfx1250) — compile-verified, matrix operand order checked by reading, not run on hardware
//
#include <hip/hip_runtime.h>

// ---------------------------------------------------------------------------
// ResVertixRefineShapenet for MI455X (gfx1250, wave32, WMMA + async LDS DMA)
// ---------------------------------------------------------------------------
#define B_SZ    4
#define N_VERT  2466
#define M_REAL  (B_SZ * N_VERT)      // 9864
#define M_PAD   9888                 // multiple of 32 (and 16)
#define GBLK    (M_PAD / 32)         // 309 GEMM blocks
#define CAP     64                   // max neighbors kept per row (mean ~7.4)
#define K0PAD   320                  // 259 padded up to multiple of 64
#define MT_ROWS 32
#define KCH     64
#define LDS_STR 72                   // ushort stride: 144B rows, conflict-free

typedef __bf16 bf16_t;
typedef __attribute__((ext_vector_type(8)))  __bf16 v8bf;
typedef __attribute__((ext_vector_type(16))) __bf16 v16bf;
typedef __attribute__((ext_vector_type(8)))  float  v8f;

__device__ __forceinline__ unsigned short f2bf(float f) {
    unsigned u = __float_as_uint(f);
    unsigned r = u + 0x7FFFu + ((u >> 16) & 1u);   // round-to-nearest-even
    return (unsigned short)(r >> 16);
}
__device__ __forceinline__ float bf2f(unsigned short h) {
    return __uint_as_float(((unsigned)h) << 16);
}
__device__ __forceinline__ v16bf combine16(v8bf lo, v8bf hi) {
    v16bf r;
#pragma unroll
    for (int i = 0; i < 8; ++i) { r[i] = lo[i]; r[i + 8] = hi[i]; }
    return r;
}
__device__ __forceinline__ void wait_async0() {
#if defined(__has_builtin) && __has_builtin(__builtin_amdgcn_s_wait_asynccnt)
    __builtin_amdgcn_s_wait_asynccnt(0);
#else
    asm volatile("s_wait_asynccnt 0" ::: "memory");
#endif
}

// ---------------------------------------------------------------------------
// NCHW (f32) -> NHWC (bf16) tiled transpose for the CNN feature maps
// ---------------------------------------------------------------------------
__global__ void nchw_to_nhwc_bf16(const float* __restrict__ src,
                                  unsigned short* __restrict__ dst,
                                  int C, int P) {
    __shared__ float tile[32][33];
    int b  = blockIdx.z;
    int c0 = blockIdx.y * 32;
    int p0 = blockIdx.x * 32;
    int tx = threadIdx.x, ty = threadIdx.y;
    const float* s = src + (size_t)b * C * P;
#pragma unroll
    for (int i = 0; i < 32; i += 8) {
        int c = c0 + ty + i, p = p0 + tx;
        tile[ty + i][tx] = (p < P) ? s[(size_t)c * P + p] : 0.0f;
    }
    __syncthreads();
    unsigned short* d = dst + (size_t)b * P * C;
#pragma unroll
    for (int i = 0; i < 32; i += 8) {
        int p = p0 + ty + i, c = c0 + tx;
        if (p < P) d[(size_t)p * C + c] = f2bf(tile[tx][ty + i]);
    }
}

// ---------------------------------------------------------------------------
// Sparsify adjacency: one wave per row; ballot+prefix keeps column order.
// Reads the 97MB adjacency exactly once.
// ---------------------------------------------------------------------------
__global__ void build_nbr(const float* __restrict__ adj,
                          int* __restrict__ cnt,
                          int* __restrict__ idx,
                          float* __restrict__ val,
                          int Mreal, int Nverts) {
    int m = blockIdx.x * (blockDim.x >> 5) + (threadIdx.x >> 5);
    if (m >= Mreal) return;
    int lane = threadIdx.x & 31;
    const float* row = adj + (size_t)m * Nverts;
    int base = 0;
    for (int c0 = 0; c0 < Nverts; c0 += 32) {
        int c = c0 + lane;
        float v = (c < Nverts) ? row[c] : 0.0f;
        unsigned mask = (unsigned)__ballot(v != 0.0f);
        int pre = __popc(mask & ((1u << lane) - 1u));
        if (v != 0.0f) {
            int p = base + pre;
            if (p < CAP) {
                idx[(size_t)m * CAP + p] = c;
                val[(size_t)m * CAP + p] = v;
            }
        }
        base += __popc(mask);
    }
    if (lane == 0) cnt[m] = (base < CAP) ? base : CAP;
}

// ---------------------------------------------------------------------------
// Weight convert: W[K,128] f32 -> Wt[128,Kpad] bf16 (transposed, zero padded)
// ---------------------------------------------------------------------------
__global__ void wt_convert(const float* __restrict__ W, int K, int Kpad,
                           unsigned short* __restrict__ Wt) {
    int t = blockIdx.x * blockDim.x + threadIdx.x;
    if (t >= 128 * Kpad) return;
    int n = t / Kpad, k = t % Kpad;
    Wt[(size_t)n * Kpad + k] = (k < K) ? f2bf(W[(size_t)k * 128 + n]) : (unsigned short)0;
}

// ---------------------------------------------------------------------------
// Vertex align: one wave per vertex; lanes parallel over channels.
// Reproduces the reference's integer-cast "bilinear" weights exactly.
// ---------------------------------------------------------------------------
__global__ void vertex_align(const float* __restrict__ pos,
                             const unsigned short* __restrict__ ft0,
                             const unsigned short* __restrict__ ft1,
                             const unsigned short* __restrict__ ft2,
                             const unsigned short* __restrict__ ft3,
                             unsigned short* __restrict__ aligned,
                             int Mreal, int Nverts) {
    int m = blockIdx.x * (blockDim.x >> 5) + (threadIdx.x >> 5);
    if (m >= Mreal) return;
    int lane = threadIdx.x & 31;
    int b = m / Nverts;
    float px = pos[(size_t)m * 3 + 0];
    float py = pos[(size_t)m * 3 + 1];
    float pz = pos[(size_t)m * 3 + 2];
    float h = fminf(fmaxf(248.0f * (py / pz) + 111.5f, 0.0f), 223.0f);
    float w = fminf(fmaxf(248.0f * (px / (-pz)) + 111.5f, 0.0f), 223.0f);

    const unsigned short* fts[4] = {ft0, ft1, ft2, ft3};
    const int Ss[4]   = {56, 28, 14, 7};
    const int Cs[4]   = {256, 512, 1024, 2048};
    const int offs[4] = {0, 256, 768, 1792};
    unsigned short* arow = aligned + (size_t)m * 3840;

    for (int mp = 0; mp < 4; ++mp) {
        int S = Ss[mp], C = Cs[mp];
        float scale = (float)S / 224.0f;           // exact power of two
        float x = h * scale, y = w * scale;
        float fx1 = floorf(x), fx2 = fminf(ceilf(x), (float)(S - 1));
        float fy1 = floorf(y), fy2 = fminf(ceilf(y), (float)(S - 1));
        int x1 = (int)fx1, x2 = (int)fx2, xi = (int)x;
        int y1 = (int)fy1, y2 = (int)fy2, yi = (int)y;
        float w11 = (float)((x2 - xi) * (y2 - yi));
        float w12 = (float)((x2 - xi) * (yi - y1));
        float w21 = (float)((xi - x1) * (y2 - yi));
        float w22 = (float)((xi - x1) * (yi - y1));
        const unsigned short* f = fts[mp] + (size_t)b * S * S * C;
        const unsigned short* p11 = f + ((size_t)x1 * S + y1) * C;
        const unsigned short* p21 = f + ((size_t)x2 * S + y1) * C;
        const unsigned short* p12 = f + ((size_t)x1 * S + y2) * C;
        const unsigned short* p22 = f + ((size_t)x2 * S + y2) * C;
        for (int c0 = 0; c0 < C; c0 += 32) {
            int c = c0 + lane;
            float acc = 0.0f;
            if (w11 != 0.0f) acc += w11 * bf2f(p11[c]);
            if (w21 != 0.0f) acc += w21 * bf2f(p21[c]);
            if (w12 != 0.0f) acc += w12 * bf2f(p12[c]);
            if (w22 != 0.0f) acc += w22 * bf2f(p22[c]);
            arow[offs[mp] + c] = f2bf(acc);
        }
    }
}

// ---------------------------------------------------------------------------
// Fill x0 columns 128..319: vertex_features (128), positions (3), zeros
// ---------------------------------------------------------------------------
__global__ void fill_xcat(const float* __restrict__ vf,
                          const float* __restrict__ pos,
                          unsigned short* __restrict__ x0,
                          int Mreal, int Mpad) {
    int t = blockIdx.x * blockDim.x + threadIdx.x;
    if (t >= Mpad * 192) return;
    int m = t / 192, c = t % 192;
    float v = 0.0f;
    if (m < Mreal) {
        if (c < 128)       v = vf[(size_t)m * 128 + c];
        else if (c < 131)  v = pos[(size_t)m * 3 + (c - 128)];
    }
    x0[(size_t)m * K0PAD + 128 + c] = f2bf(v);
}

// ---------------------------------------------------------------------------
// WMMA GEMM:  out = [relu]( A[M,Kpad] @ Wt^T + bias + add1 ) + add2
//   block: 32 M-rows x 128 N-cols, 8 waves, 2 WMMA row-tiles per wave.
//   K loop in chunks of 64; A tile double-buffered in LDS, filled with
//   global_load_async_to_lds_b128 (ASYNCcnt) overlapping the WMMAs.
// ---------------------------------------------------------------------------
__global__ void __launch_bounds__(256)
gemm_wmma_bf16(const unsigned short* __restrict__ A, int lda,
               const unsigned short* __restrict__ Wt, int Kpad,
               const float* __restrict__ bias,
               const float* __restrict__ add1, int relu,
               const float* __restrict__ add2,
               float* __restrict__ outf,
               unsigned short* __restrict__ outb, int ldo, int col0) {
    __shared__ __align__(16) unsigned short As[2][MT_ROWS * LDS_STR];

    int m0   = blockIdx.x * MT_ROWS;
    int tid  = threadIdx.x;
    int wave = tid >> 5;
    int lane = tid & 31;
    int n0   = wave * 16;
    int rowA = lane & 15;
    int hi   = (lane >> 4) & 1;
    int koff = hi * 8;            // K sub-offset per ISA A/B fragment layout

    // async-fill mapping: 256 threads x 16B cover one 32x64 bf16 tile
    int fr = tid >> 3;            // 0..31 tile row
    int fc = (tid & 7) * 8;       // 0..56 col (elements)
    const unsigned short* gsrc = A + (size_t)(m0 + fr) * lda + fc;
    unsigned lds0 = (unsigned)(uintptr_t)(&As[0][fr * LDS_STR + fc]);
    unsigned lds1 = (unsigned)(uintptr_t)(&As[1][fr * LDS_STR + fc]);

    v8f acc0 = {0.f, 0.f, 0.f, 0.f, 0.f, 0.f, 0.f, 0.f};
    v8f acc1 = {0.f, 0.f, 0.f, 0.f, 0.f, 0.f, 0.f, 0.f};

    // prologue: fill buffer 0 for k=0
    {
        unsigned long long ga = (unsigned long long)(uintptr_t)gsrc;
        asm volatile("global_load_async_to_lds_b128 %0, %1, off"
                     :: "v"(lds0), "v"(ga) : "memory");
    }
    wait_async0();
    __syncthreads();

    int nch = Kpad / KCH;
    for (int ch = 0; ch < nch; ++ch) {
        int cur = ch & 1;
        // prefetch next chunk into the other buffer (overlaps the WMMAs)
        if (ch + 1 < nch) {
            unsigned long long ga =
                (unsigned long long)(uintptr_t)(gsrc + (size_t)(ch + 1) * KCH);
            asm volatile("global_load_async_to_lds_b128 %0, %1, off"
                         :: "v"(cur ? lds0 : lds1), "v"(ga) : "memory");
        }
        int k0 = ch * KCH;
        const unsigned short* asb = cur ? &As[1][0] : &As[0][0];
#pragma unroll
        for (int kk = 0; kk < KCH; kk += 32) {
            const unsigned short* wrow =
                Wt + (size_t)(n0 + rowA) * Kpad + k0 + kk + koff;
            v16bf bfrag = combine16(
                *reinterpret_cast<const v8bf*>(wrow),
                *reinterpret_cast<const v8bf*>(wrow + 16));
            {
                const unsigned short* as0 = asb + rowA * LDS_STR + kk + koff;
                v16bf afrag = combine16(
                    *reinterpret_cast<const v8bf*>(as0),
                    *reinterpret_cast<const v8bf*>(as0 + 16));
                acc0 = __builtin_amdgcn_wmma_f32_16x16x32_bf16(
                    false, afrag, false, bfrag, (short)0, acc0, false, false);
            }
            {
                const unsigned short* as1 =
                    asb + (16 + rowA) * LDS_STR + kk + koff;
                v16bf afrag = combine16(
                    *reinterpret_cast<const v8bf*>(as1),
                    *reinterpret_cast<const v8bf*>(as1 + 16));
                acc1 = __builtin_amdgcn_wmma_f32_16x16x32_bf16(
                    false, afrag, false, bfrag, (short)0, acc1, false, false);
            }
        }
        wait_async0();
        __syncthreads();
    }

    // D layout: element g -> (M = m0 + t*16 + g + hi*8, N = n0 + (lane&15))
#pragma unroll
    for (int t = 0; t < 2; ++t) {
        v8f acc = t ? acc1 : acc0;
#pragma unroll
        for (int g = 0; g < 8; ++g) {
            int m = m0 + t * 16 + g + hi * 8;
            int n = n0 + rowA;
            float v = acc[g];
            if (bias) v += bias[n];
            if (add1) v += add1[(size_t)m * 128 + n];
            if (relu) v = fmaxf(v, 0.0f);
            if (add2) v += add2[(size_t)m * 128 + n];
            if (outf) outf[(size_t)m * 128 + n] = v;
            if (outb) outb[(size_t)m * ldo + col0 + n] = f2bf(v);
        }
    }
}

// ---------------------------------------------------------------------------
// SpMM: t2[m,:] = sum_j val * t1[b*N + idx, :]   (128 channels, 1 block/row)
// ---------------------------------------------------------------------------
__global__ void spmm128(const float* __restrict__ t1,
                        const int* __restrict__ cnt,
                        const int* __restrict__ idx,
                        const float* __restrict__ val,
                        float* __restrict__ t2,
                        int Mreal, int Nverts) {
    int m = blockIdx.x;
    if (m >= Mreal) return;
    int c = threadIdx.x;
    int b = m / Nverts;
    size_t base = (size_t)b * Nverts;
    int nn = cnt[m];
    float acc = 0.0f;
    for (int j = 0; j < nn; ++j) {
        int   jj = idx[(size_t)m * CAP + j];
        float w  = val[(size_t)m * CAP + j];
        acc += w * t1[(base + jj) * 128 + c];
    }
    t2[(size_t)m * 128 + c] = acc;
}

// ---------------------------------------------------------------------------
// Final head: uv[m,0..2] = x@gf_w0 ; uv[m,3..5] = x@gf_w1
// ---------------------------------------------------------------------------
__global__ void gf_proj(const unsigned short* __restrict__ x,
                        const float* __restrict__ w0,
                        const float* __restrict__ w1,
                        float* __restrict__ uv, int Mreal) {
    int m = blockIdx.x * blockDim.x + threadIdx.x;
    if (m >= Mreal) return;
    float a0 = 0, a1 = 0, a2 = 0, b0 = 0, b1 = 0, b2 = 0;
    const unsigned short* xr = x + (size_t)m * 128;
    for (int k = 0; k < 128; ++k) {
        float xv = bf2f(xr[k]);
        a0 += xv * w0[k * 3 + 0]; a1 += xv * w0[k * 3 + 1]; a2 += xv * w0[k * 3 + 2];
        b0 += xv * w1[k * 3 + 0]; b1 += xv * w1[k * 3 + 1]; b2 += xv * w1[k * 3 + 2];
    }
    float* o = uv + (size_t)m * 6;
    o[0] = a0; o[1] = a1; o[2] = a2; o[3] = b0; o[4] = b1; o[5] = b2;
}

__global__ void spmm3(const float* __restrict__ uv,
                      const int* __restrict__ cnt,
                      const int* __restrict__ idx,
                      const float* __restrict__ val,
                      float* __restrict__ t2g, int Mreal, int Nverts) {
    int m = blockIdx.x * blockDim.x + threadIdx.x;
    if (m >= Mreal) return;
    int b = m / Nverts;
    size_t base = (size_t)b * Nverts;
    int nn = cnt[m];
    float a0 = 0, a1 = 0, a2 = 0;
    for (int j = 0; j < nn; ++j) {
        int   jj = idx[(size_t)m * CAP + j];
        float w  = val[(size_t)m * CAP + j];
        const float* r = uv + (base + jj) * 6 + 3;
        a0 += w * r[0]; a1 += w * r[1]; a2 += w * r[2];
    }
    float* o = t2g + (size_t)m * 3;
    o[0] = a0; o[1] = a1; o[2] = a2;
}

__global__ void gf_final(const float* __restrict__ uv,
                         const float* __restrict__ t2g,
                         const float* __restrict__ pos,
                         float* __restrict__ out, int Mreal) {
    int t = blockIdx.x * blockDim.x + threadIdx.x;
    if (t >= Mreal * 3) return;
    int m = t / 3, j = t % 3;
    float d = uv[(size_t)m * 6 + j] + t2g[(size_t)m * 3 + j];
    d = fmaxf(d, 0.0f);
    out[t] = pos[t] + tanhf(d);
}

// ---------------------------------------------------------------------------
extern "C" void kernel_launch(void* const* d_in, const int* in_sizes, int n_in,
                              void* d_out, int out_size, void* d_ws, size_t ws_size,
                              hipStream_t stream) {
    (void)in_sizes; (void)n_in; (void)out_size; (void)ws_size;

    const float* conv[4] = {(const float*)d_in[0], (const float*)d_in[1],
                            (const float*)d_in[2], (const float*)d_in[3]};
    const float* pos  = (const float*)d_in[4];
    const float* vf   = (const float*)d_in[5];
    const float* adj  = (const float*)d_in[6];
    const float* lin_w = (const float*)d_in[7];
    const float* lin_b = (const float*)d_in[8];
    const float* gfw0 = (const float*)d_in[27];
    const float* gfw1 = (const float*)d_in[28];

    // ---- workspace layout -------------------------------------------------
    char* p = (char*)d_ws;
    auto alloc = [&](size_t bytes) -> char* {
        char* r = p;
        p += (bytes + 255) & ~(size_t)255;
        return r;
    };
    const int Ss[4] = {56, 28, 14, 7};
    const int Cs[4] = {256, 512, 1024, 2048};
    unsigned short* ft[4];
    for (int i = 0; i < 4; ++i)
        ft[i] = (unsigned short*)alloc((size_t)B_SZ * Cs[i] * Ss[i] * Ss[i] * 2);
    unsigned short* aligned = (unsigned short*)alloc((size_t)M_PAD * 3840 * 2);
    unsigned short* lin_wt  = (unsigned short*)alloc((size_t)128 * 3840 * 2);
    unsigned short* wt[3][5];
    const int Kin[3] = {259, 128, 128};
    const int Kpd[3] = {K0PAD, 128, 128};
    for (int j = 0; j < 3; ++j) {
        wt[j][0] = (unsigned short*)alloc((size_t)128 * Kpd[j] * 2);  // pw
        wt[j][1] = (unsigned short*)alloc((size_t)128 * Kpd[j] * 2);  // w00
        wt[j][2] = (unsigned short*)alloc((size_t)128 * Kpd[j] * 2);  // w01
        wt[j][3] = (unsigned short*)alloc((size_t)128 * 128 * 2);     // w10
        wt[j][4] = (unsigned short*)alloc((size_t)128 * 128 * 2);     // w11
    }
    unsigned short* x0 = (unsigned short*)alloc((size_t)M_PAD * K0PAD * 2);
    unsigned short* xb[3];
    for (int j = 0; j < 3; ++j)
        xb[j] = (unsigned short*)alloc((size_t)M_PAD * 128 * 2);
    unsigned short* h1 = (unsigned short*)alloc((size_t)M_PAD * 128 * 2);
    float* t1  = (float*)alloc((size_t)M_PAD * 128 * 4);
    float* t2  = (float*)alloc((size_t)M_PAD * 128 * 4);
    float* sb  = (float*)alloc((size_t)M_PAD * 128 * 4);
    int*   cnt = (int*)alloc((size_t)M_PAD * 4);
    int*   idx = (int*)alloc((size_t)M_PAD * CAP * 4);
    float* valv = (float*)alloc((size_t)M_PAD * CAP * 4);
    float* uv  = (float*)alloc((size_t)M_PAD * 6 * 4);
    float* t2g = (float*)alloc((size_t)M_PAD * 3 * 4);

    // ---- stage 0: feature-map transposes, adjacency sparsify, weights -----
    for (int i = 0; i < 4; ++i) {
        int P = Ss[i] * Ss[i];
        dim3 g((P + 31) / 32, Cs[i] / 32, B_SZ), b(32, 8);
        nchw_to_nhwc_bf16<<<g, b, 0, stream>>>(conv[i], ft[i], Cs[i], P);
    }
    build_nbr<<<(M_REAL + 7) / 8, 256, 0, stream>>>(adj, cnt, idx, valv,
                                                    M_REAL, N_VERT);
    wt_convert<<<(128 * 3840 + 255) / 256, 256, 0, stream>>>(lin_w, 3840, 3840,
                                                             lin_wt);
    for (int j = 0; j < 3; ++j) {
        const float* rw[5] = {(const float*)d_in[9 + 6 * j + 0],
                              (const float*)d_in[9 + 6 * j + 2],
                              (const float*)d_in[9 + 6 * j + 3],
                              (const float*)d_in[9 + 6 * j + 4],
                              (const float*)d_in[9 + 6 * j + 5]};
        for (int q = 0; q < 5; ++q) {
            int K  = (q < 3) ? Kin[j] : 128;
            int Kp = (q < 3) ? Kpd[j] : 128;
            wt_convert<<<(128 * Kp + 255) / 256, 256, 0, stream>>>(rw[q], K, Kp,
                                                                   wt[j][q]);
        }
    }

    // ---- stage 1: vertex align + projection + concat ----------------------
    vertex_align<<<(M_REAL + 7) / 8, 256, 0, stream>>>(
        pos, ft[0], ft[1], ft[2], ft[3], aligned, M_REAL, N_VERT);

    gemm_wmma_bf16<<<GBLK, 256, 0, stream>>>(
        aligned, 3840, lin_wt, 3840, lin_b,
        nullptr, 0, nullptr, nullptr, x0, K0PAD, 0);

    fill_xcat<<<((size_t)M_PAD * 192 + 255) / 256, 256, 0, stream>>>(
        vf, pos, x0, M_REAL, M_PAD);

    // ---- stage 2: three residual graph-conv blocks ------------------------
    const unsigned short* xin = x0;
    int lda = K0PAD, Kp = K0PAD;
    for (int j = 0; j < 3; ++j) {
        const float* pb = (const float*)d_in[9 + 6 * j + 1];
        // skip = x @ pw + pb
        gemm_wmma_bf16<<<GBLK, 256, 0, stream>>>(
            xin, lda, wt[j][0], Kp, pb, nullptr, 0, nullptr, sb, nullptr, 0, 0);
        // t1 = x @ w01 ; t2 = A @ t1
        gemm_wmma_bf16<<<GBLK, 256, 0, stream>>>(
            xin, lda, wt[j][2], Kp, nullptr, nullptr, 0, nullptr, t1,
            nullptr, 0, 0);
        spmm128<<<M_REAL, 128, 0, stream>>>(t1, cnt, idx, valv, t2, M_REAL, N_VERT);
        // h1 = relu(x @ w00 + t2)
        gemm_wmma_bf16<<<GBLK, 256, 0, stream>>>(
            xin, lda, wt[j][1], Kp, nullptr, t2, 1, nullptr, nullptr, h1, 128, 0);
        // t1 = h1 @ w11 ; t2 = A @ t1
        gemm_wmma_bf16<<<GBLK, 256, 0, stream>>>(
            h1, 128, wt[j][4], 128, nullptr, nullptr, 0, nullptr, t1,
            nullptr, 0, 0);
        spmm128<<<M_REAL, 128, 0, stream>>>(t1, cnt, idx, valv, t2, M_REAL, N_VERT);
        // x_next = skip + relu(h1 @ w10 + t2)
        gemm_wmma_bf16<<<GBLK, 256, 0, stream>>>(
            h1, 128, wt[j][3], 128, nullptr, t2, 1, sb, nullptr, xb[j], 128, 0);
        xin = xb[j];
        lda = 128; Kp = 128;
    }

    // ---- stage 3: 3-channel head ------------------------------------------
    gf_proj<<<(M_REAL + 127) / 128, 128, 0, stream>>>(xb[2], gfw0, gfw1, uv, M_REAL);
    spmm3<<<(M_REAL + 255) / 256, 256, 0, stream>>>(uv, cnt, idx, valv, t2g,
                                                    M_REAL, N_VERT);
    gf_final<<<(M_REAL * 3 + 255) / 256, 256, 0, stream>>>(
        uv, t2g, pos, (float*)d_out, M_REAL);
}
